// Bo_27573690040642
// MI455X (gfx1250) — matrix operand order repaired and checked by reading, not compiled, not run on hardware
//
#include <hip/hip_runtime.h>
#include <hip/hip_bf16.h>

typedef __attribute__((ext_vector_type(16))) _Float16 v16h;
typedef __attribute__((ext_vector_type(8)))  float    v8f;

struct __align__(16) h8 { _Float16 v[8]; };

#define WMMA_F16(a,b,c) __builtin_amdgcn_wmma_f32_16x16x32_f16(false,(a),false,(b),(short)0,(c),false,false)

__device__ __forceinline__ float leakyf(float x){ return x >= 0.f ? x : 0.2f*x; }

__device__ __forceinline__ void ld8lo(v16h& a, const _Float16* p){
  const h8 r = *(const h8*)p;
#pragma unroll
  for (int j = 0; j < 8; ++j) a[j] = r.v[j];
}
__device__ __forceinline__ void ld8hi(v16h& a, const _Float16* p){
  const h8 r = *(const h8*)p;
#pragma unroll
  for (int j = 0; j < 8; ++j) a[8+j] = r.v[j];
}

// ---------------- sizes ----------------
#define PIX0 131072   // 8*128*128
#define PIX1 524288   // 8*256*256

// ---------------- ws layout (bytes) ----------------
constexpr size_t OFF_XHALF  = 0;                        // 16777216
constexpr size_t OFF_Y1     = OFF_XHALF + 16777216;     // 4194304
constexpr size_t OFF_Y2     = OFF_Y1    + 4194304;      // 4194304
constexpr size_t OFF_Y256   = OFF_Y2    + 4194304;      // 16777216
constexpr size_t OFF_Z1     = OFF_Y256  + 16777216;     // 16777216
constexpr size_t OFF_Z2     = OFF_Z1    + 16777216;     // 16777216
constexpr size_t OFF_OUTPRE = OFF_Z2    + 16777216;     // 134217728
constexpr size_t OFF_TN     = OFF_OUTPRE+ 134217728;    // 67108864
constexpr size_t OFF_W7A    = OFF_TN    + 67108864;     // 100352
constexpr size_t OFF_W1     = OFF_W7A   + 100352;       // 1024
constexpr size_t OFF_W3A    = OFF_W1    + 1024;         // 5120
constexpr size_t OFF_W3B    = OFF_W3A   + 5120;
constexpr size_t OFF_W3C    = OFF_W3B   + 5120;
constexpr size_t OFF_W7B    = OFF_W3C   + 5120;         // 102400
constexpr size_t OFF_WFC1   = OFF_W7B   + 102400;       // 8192
constexpr size_t OFF_WFC2   = OFF_WFC1  + 8192;
constexpr size_t OFF_PMAX   = OFF_WFC2  + 8192;         // 8192
constexpr size_t OFF_KMAX   = OFF_PMAX  + 8192;         // 32 (padded)
constexpr size_t OFF_PSE    = OFF_KMAX  + 256;          // 8192
constexpr size_t OFF_PSV    = OFF_PSE   + 8192;         // 65536
constexpr size_t OFF_ADDV   = OFF_PSV   + 65536;        // 2048

// ================= prep kernels =================
__global__ void k_xhalf(const float* __restrict__ x, _Float16* __restrict__ xh){
  int t = blockIdx.x*256 + threadIdx.x;          // < PIX0*64
  int c = t & 63; int pix = t >> 6;
  int w = pix & 127, h = (pix >> 7) & 127, n = pix >> 14;
  xh[t] = (_Float16)x[(((size_t)(n*64 + c)*128 + h)*128) + w];
}

// B layout: dst[(chunk*32+lane)*16 + j] = Wg[k=(j/8)*16+(lane/16)*8+(j%8) + chunk*32][n=lane&15]
__global__ void k_pack7a(const float* __restrict__ w, _Float16* __restrict__ dst){
  int t = blockIdx.x*256 + threadIdx.x;          // < 98*512
  int chunk = t >> 9; int r = t & 511;
  int lane = r >> 4, j = r & 15;
  int klocal = ((j >> 3) << 4) + ((lane >> 4) << 3) + (j & 7);
  int n = lane & 15;
  int kk = chunk*32 + klocal;                    // kk = tap*64 + ch
  int tap = kk >> 6, ch = kk & 63;
  int ky = tap/7, kx = tap%7;
  dst[t] = (_Float16)w[(((size_t)(n*64 + ch)*7 + ky)*7) + kx];
}

__global__ void k_pack1(const float* __restrict__ w, _Float16* __restrict__ dst){
  int t = blockIdx.x*256 + threadIdx.x; if (t >= 512) return;
  int lane = t >> 4, j = t & 15;
  int klocal = ((j >> 3) << 4) + ((lane >> 4) << 3) + (j & 7);
  int n = lane & 15;
  float v = (klocal < 16) ? w[n*16 + klocal] : 0.f;
  dst[t] = (_Float16)v;
}

__global__ void k_pack3(const float* __restrict__ w, _Float16* __restrict__ dst){
  int t = blockIdx.x*256 + threadIdx.x;          // < 5*512
  int chunk = t >> 9; int r = t & 511;
  int lane = r >> 4, j = r & 15;
  int klocal = ((j >> 3) << 4) + ((lane >> 4) << 3) + (j & 7);
  int n = lane & 15;
  int kk = chunk*32 + klocal;                    // kk = tap*16 + ch
  int tap = kk >> 4, ch = kk & 15;
  float v = 0.f;
  if (tap < 9){ int ky = tap/3, kx = tap%3; v = w[(((size_t)(n*16 + ch)*3 + ky)*3) + kx]; }
  dst[t] = (_Float16)v;
}

__global__ void k_pack7b(const float* __restrict__ w, _Float16* __restrict__ dst){
  int t = blockIdx.x*256 + threadIdx.x;          // < 4*25*512
  int nt = t / 12800; int r0 = t % 12800;
  int chunk = r0 >> 9; int r = r0 & 511;
  int lane = r >> 4, j = r & 15;
  int klocal = ((j >> 3) << 4) + ((lane >> 4) << 3) + (j & 7);
  int n = (lane & 15) + nt*16;
  int kk = chunk*32 + klocal;
  int tap = kk >> 4, ch = kk & 15;
  float v = 0.f;
  if (tap < 49){ int ky = tap/7, kx = tap%7; v = w[(((size_t)(n*16 + ch)*7 + ky)*7) + kx]; }
  dst[t] = (_Float16)v;
}

__global__ void k_packfc(const float* __restrict__ w, _Float16* __restrict__ dst){
  int t = blockIdx.x*256 + threadIdx.x;          // < 4096
  int nt = t >> 10; int r0 = t & 1023;
  int chunk = r0 >> 9; int r = r0 & 511;
  int lane = r >> 4, j = r & 15;
  int klocal = ((j >> 3) << 4) + ((lane >> 4) << 3) + (j & 7);
  int o = (lane & 15) + nt*16;
  int ch = chunk*32 + klocal;
  dst[t] = (_Float16)w[o*64 + ch];               // B[k][n] = W[n_out][k_in]
}

// ================= WMMA conv kernels =================
// 7x7 pad3, 64->16 @128^2, leaky. Interior fast path (uniform branch), guarded border path.
__global__ void k_conv7a(const _Float16* __restrict__ xh, const _Float16* __restrict__ wp,
                         const float* __restrict__ bias, _Float16* __restrict__ y1){
  int wid = (blockIdx.x*256 + threadIdx.x) >> 5;
  int lane = threadIdx.x & 31, hi = lane >> 4;
  int pbase = wid << 4;
  int p = pbase + (lane & 15);
  int w = p & 127, h = (p >> 7) & 127, n = p >> 14;
  v8f acc = {};
  int wb = pbase & 127;
  int interior = __builtin_amdgcn_readfirstlane(
      (h >= 3 && h <= 124 && wb >= 16 && wb <= 96) ? 1 : 0);
  if (interior){
    const _Float16* pp = xh + (size_t)(p - 387)*64 + hi*8;   // -3*128-3
    const _Float16* wq = wp + lane*16;
    for (int ky = 0; ky < 7; ++ky){
      for (int kx = 0; kx < 7; ++kx){
#pragma unroll
        for (int cp = 0; cp < 2; ++cp){
          v16h a;
          ld8lo(a, pp + cp*32);
          ld8hi(a, pp + cp*32 + 16);
          v16h b = *(const v16h*)wq; wq += 512;
          acc = WMMA_F16(a, b, acc);
        }
        pp += 64;
      }
      pp += (128 - 7)*64;
    }
  } else {
    int ty = -3, tx = -3;
    for (int tap = 0; tap < 49; ++tap){
      int ih = h + ty, iw = w + tx;
      bool inb = (unsigned)ih < 128u && (unsigned)iw < 128u;
      const _Float16* pp = xh + (size_t)(n*16384 + ih*128 + iw)*64 + hi*8;
#pragma unroll
      for (int cp = 0; cp < 2; ++cp){
        v16h a = {};
        if (inb){ ld8lo(a, pp + cp*32); ld8hi(a, pp + cp*32 + 16); }
        v16h b = *(const v16h*)(wp + (((tap*2 + cp) << 5) + lane)*16);
        acc = WMMA_F16(a, b, acc);
      }
      ++tx; if (tx > 3){ tx = -3; ++ty; }
    }
  }
  int cout = lane & 15;
  float bi = bias[cout];
#pragma unroll
  for (int r = 0; r < 8; ++r){
    float v = leakyf(acc[r] + bi);
    y1[(size_t)(pbase + hi*8 + r)*16 + cout] = (_Float16)v;
  }
}

// 1x1 16->16 + leaky + BN @128^2 (K padded to 32 with zeros)
__global__ void k_conv1bn(const _Float16* __restrict__ y1, const _Float16* __restrict__ wp,
                          const float* __restrict__ cb, const float* __restrict__ bng,
                          const float* __restrict__ bnb, _Float16* __restrict__ y2){
  int wid = (blockIdx.x*256 + threadIdx.x) >> 5;
  int lane = threadIdx.x & 31, hi = lane >> 4;
  int pbase = wid << 4;
  int p = pbase + (lane & 15);
  v16h a = {};
  ld8lo(a, y1 + (size_t)p*16 + hi*8);
  v16h b = *(const v16h*)(wp + lane*16);
  v8f acc = {};
  acc = WMMA_F16(a, b, acc);
  int cout = lane & 15;
  float scale = bng[cout]*rsqrtf(1.f + 1e-5f);
  float bb = bnb[cout], bi = cb[cout];
#pragma unroll
  for (int r = 0; r < 8; ++r){
    float v = leakyf(acc[r] + bi);
    v = v*scale + bb;
    y2[(size_t)(pbase + hi*8 + r)*16 + cout] = (_Float16)v;
  }
}

// bilinear 128->256 align_corners, 16ch NHWC f16
__global__ void k_resize16(const _Float16* __restrict__ src, _Float16* __restrict__ dst){
  int pid = blockIdx.x*256 + threadIdx.x;        // < PIX1
  int ow = pid & 255, oh = (pid >> 8) & 255, n = pid >> 16;
  const float s = 127.f/255.f;
  float sy = oh*s, sx = ow*s;
  int y0 = (int)sy; if (y0 > 126) y0 = 126;
  int x0 = (int)sx; if (x0 > 126) x0 = 126;
  float fy = sy - y0, fx = sx - x0;
  const _Float16* p00 = src + (size_t)(n*16384 + y0*128 + x0)*16;
  const _Float16* p10 = p00 + 128*16;
  _Float16* q = dst + (size_t)pid*16;
#pragma unroll
  for (int c = 0; c < 16; ++c){
    float top = (float)p00[c]*(1.f - fx) + (float)p00[16 + c]*fx;
    float bot = (float)p10[c]*(1.f - fx) + (float)p10[16 + c]*fx;
    q[c] = (_Float16)(top*(1.f - fy) + bot*fy);
  }
}

// generic 3x3 pad1 16->16 @256^2, +BN (+leaky) (+residual). Interior fast path.
__global__ void k_conv3(const _Float16* __restrict__ in, const _Float16* __restrict__ wp,
                        const float* __restrict__ cb, const float* __restrict__ bng,
                        const float* __restrict__ bnb, const _Float16* __restrict__ resid,
                        int do_leaky, _Float16* __restrict__ out){
  int wid = (blockIdx.x*256 + threadIdx.x) >> 5;
  int lane = threadIdx.x & 31, hi = lane >> 4;
  int pbase = wid << 4;
  int p = pbase + (lane & 15);
  int w = p & 255, h = (p >> 8) & 255, n = p >> 16;
  v8f acc = {};
  int wb = pbase & 255;
  int interior = __builtin_amdgcn_readfirstlane(
      (h >= 1 && h <= 254 && wb >= 16 && wb <= 224) ? 1 : 0);
  if (interior){
    const _Float16* base = in + (size_t)p*16 + hi*8;
    const _Float16* wq = wp + lane*16;
    int ty0 = -1, tx0 = -1;      // tap0 = 2*chunk
    int ty1 = -1, tx1 =  0;      // tap1 = 2*chunk+1
    for (int chunk = 0; chunk < 4; ++chunk){
      v16h a;
      ld8lo(a, base + (ty0*256 + tx0)*16);
      ld8hi(a, base + (ty1*256 + tx1)*16);
      v16h b = *(const v16h*)wq; wq += 512;
      acc = WMMA_F16(a, b, acc);
      tx0 += 2; if (tx0 > 1){ tx0 -= 3; ++ty0; }
      tx1 += 2; if (tx1 > 1){ tx1 -= 3; ++ty1; }
    }
    { // chunk 4: tap0 = 8 (dy=1,dx=1), tap1 padded to zero
      v16h a = {};
      ld8lo(a, base + (256 + 1)*16);
      v16h b = *(const v16h*)wq;
      acc = WMMA_F16(a, b, acc);
    }
  } else {
    int ty0 = -1, tx0 = -1;
    int ty1 = -1, tx1 =  0;
    for (int chunk = 0; chunk < 5; ++chunk){
      v16h a = {};
      { int ih = h + ty0, iw = w + tx0;
        if ((unsigned)ih < 256u && (unsigned)iw < 256u)
          ld8lo(a, in + (size_t)((n << 16) + ih*256 + iw)*16 + hi*8); }
      if (chunk < 4){
        int ih = h + ty1, iw = w + tx1;
        if ((unsigned)ih < 256u && (unsigned)iw < 256u)
          ld8hi(a, in + (size_t)((n << 16) + ih*256 + iw)*16 + hi*8); }
      v16h b = *(const v16h*)(wp + ((chunk << 5) + lane)*16);
      acc = WMMA_F16(a, b, acc);
      tx0 += 2; if (tx0 > 1){ tx0 -= 3; ++ty0; }
      tx1 += 2; if (tx1 > 1){ tx1 -= 3; ++ty1; }
    }
  }
  int cout = lane & 15;
  float scale = bng[cout]*rsqrtf(1.f + 1e-5f);
  float bb = bnb[cout], bi = cb[cout];
#pragma unroll
  for (int r = 0; r < 8; ++r){
    size_t p2 = (size_t)(pbase + hi*8 + r);
    float v = (acc[r] + bi)*scale + bb;
    if (do_leaky) v = leakyf(v);
    if (resid) v += (float)resid[p2*16 + cout];
    out[p2*16 + cout] = (_Float16)v;
  }
}

// 7x7 pad3 16->64 @256^2; interior fast path; epilogue adds bilinear identity; f32 NHWC out.
__global__ void k_conv7b(const _Float16* __restrict__ in, const _Float16* __restrict__ wp,
                         const float* __restrict__ cb, const float* __restrict__ x,
                         float* __restrict__ outp){
  int wid = (blockIdx.x*256 + threadIdx.x) >> 5;
  int lane = threadIdx.x & 31, hi = lane >> 4;
  int tile = wid >> 2, nt = wid & 3;
  int pbase = tile << 4;
  int p = pbase + (lane & 15);
  int w = p & 255, h = (p >> 8) & 255, n = p >> 16;
  v8f acc = {};
  const _Float16* wpn = wp + (size_t)nt*25*512;
  int wb = pbase & 255;
  int interior = __builtin_amdgcn_readfirstlane(
      (h >= 3 && h <= 252 && wb >= 16 && wb <= 224) ? 1 : 0);
  if (interior){
    const _Float16* base = in + (size_t)p*16 + hi*8;
    const _Float16* wq = wpn + lane*16;
    int ty0 = -3, tx0 = -3;      // tap0 = 2*chunk
    int ty1 = -3, tx1 = -2;      // tap1 = 2*chunk+1
    for (int chunk = 0; chunk < 24; ++chunk){
      v16h a;
      ld8lo(a, base + (ty0*256 + tx0)*16);
      ld8hi(a, base + (ty1*256 + tx1)*16);
      if (chunk + 1 < 24) __builtin_prefetch(wq + 512, 0, 3);
      v16h b = *(const v16h*)wq; wq += 512;
      acc = WMMA_F16(a, b, acc);
      tx0 += 2; if (tx0 > 3){ tx0 -= 7; ++ty0; }
      tx1 += 2; if (tx1 > 3){ tx1 -= 7; ++ty1; }
    }
    { // chunk 24: tap0 = 48 (dy=3,dx=3), tap1 padded to zero
      v16h a = {};
      ld8lo(a, base + (3*256 + 3)*16);
      v16h b = *(const v16h*)wq;
      acc = WMMA_F16(a, b, acc);
    }
  } else {
    int ty0 = -3, tx0 = -3;
    int ty1 = -3, tx1 = -2;
    for (int chunk = 0; chunk < 25; ++chunk){
      v16h a = {};
      { int ih = h + ty0, iw = w + tx0;
        if ((unsigned)ih < 256u && (unsigned)iw < 256u)
          ld8lo(a, in + (size_t)((n << 16) + ih*256 + iw)*16 + hi*8); }
      if (chunk < 24){
        int ih = h + ty1, iw = w + tx1;
        if ((unsigned)ih < 256u && (unsigned)iw < 256u)
          ld8hi(a, in + (size_t)((n << 16) + ih*256 + iw)*16 + hi*8); }
      v16h b = *(const v16h*)(wpn + ((chunk << 5) + lane)*16);
      acc = WMMA_F16(a, b, acc);
      tx0 += 2; if (tx0 > 3){ tx0 -= 7; ++ty0; }
      tx1 += 2; if (tx1 > 3){ tx1 -= 7; ++ty1; }
    }
  }
  int cout = nt*16 + (lane & 15);
  float bi = cb[cout];
  const float s = 127.f/255.f;
  int oh = (pbase >> 8) & 255, nb = pbase >> 16;
  float sy = oh*s; int y0 = (int)sy; if (y0 > 126) y0 = 126; float fy = sy - y0;
  const float* xc = x + ((size_t)(nb*64 + cout)*128 + y0)*128;
#pragma unroll
  for (int r = 0; r < 8; ++r){
    int m = hi*8 + r;
    int ow = (pbase & 255) + m;
    float sx = ow*s; int x0 = (int)sx; if (x0 > 126) x0 = 126; float fx = sx - x0;
    float t0 = xc[x0]*(1.f - fx) + xc[x0 + 1]*fx;
    float t1 = xc[128 + x0]*(1.f - fx) + xc[128 + x0 + 1]*fx;
    outp[(size_t)(pbase + m)*64 + cout] = acc[r] + bi + t0*(1.f - fy) + t1*fy;
  }
}

// ================= attention (collapses to per-batch bias) =================
__global__ void k_att1(const float* __restrict__ outp, const float* __restrict__ wk,
                       const float* __restrict__ bk, float* __restrict__ partmax){
  int n = blockIdx.x >> 8, b = blockIdx.x & 255;
  int p = (n << 16) + (b << 8) + threadIdx.x;
  const float* t = outp + (size_t)p*64;
  float kd = bk[0];
  for (int c = 0; c < 64; ++c) kd += t[c]*wk[c];
  __shared__ float sm[256];
  sm[threadIdx.x] = kd; __syncthreads();
  for (int s = 128; s > 0; s >>= 1){
    if (threadIdx.x < s) sm[threadIdx.x] = fmaxf(sm[threadIdx.x], sm[threadIdx.x + s]);
    __syncthreads();
  }
  if (threadIdx.x == 0) partmax[blockIdx.x] = sm[0];
}

__global__ void k_att2(const float* __restrict__ partmax, float* __restrict__ kmax){
  int n = blockIdx.x;
  __shared__ float sm[256];
  sm[threadIdx.x] = partmax[n*256 + threadIdx.x]; __syncthreads();
  for (int s = 128; s > 0; s >>= 1){
    if (threadIdx.x < s) sm[threadIdx.x] = fmaxf(sm[threadIdx.x], sm[threadIdx.x + s]);
    __syncthreads();
  }
  if (threadIdx.x == 0) kmax[n] = sm[0];
}

__global__ void k_att3(const float* __restrict__ outp, const float* __restrict__ wk,
                       const float* __restrict__ bk, const float* __restrict__ wv,
                       const float* __restrict__ bv, const float* __restrict__ kmax,
                       float* __restrict__ partse, float* __restrict__ partsv){
  int n = blockIdx.x >> 8, b = blockIdx.x & 255;
  int p = (n << 16) + (b << 8) + threadIdx.x;
  const float* t = outp + (size_t)p*64;
  float kd = bk[0]; float vd[8];
#pragma unroll
  for (int j = 0; j < 8; ++j) vd[j] = bv[j];
  for (int c = 0; c < 64; ++c){
    float tc = t[c];
    kd += tc*wk[c];
#pragma unroll
    for (int j = 0; j < 8; ++j) vd[j] += tc*wv[j*64 + c];
  }
  float e = expf(kd - kmax[n]);
  __shared__ float sm[256];
  sm[threadIdx.x] = e; __syncthreads();
  for (int s = 128; s > 0; s >>= 1){
    if (threadIdx.x < s) sm[threadIdx.x] += sm[threadIdx.x + s];
    __syncthreads();
  }
  if (threadIdx.x == 0) partse[blockIdx.x] = sm[0];
  __syncthreads();
  for (int j = 0; j < 8; ++j){
    sm[threadIdx.x] = e*vd[j]; __syncthreads();
    for (int s = 128; s > 0; s >>= 1){
      if (threadIdx.x < s) sm[threadIdx.x] += sm[threadIdx.x + s];
      __syncthreads();
    }
    if (threadIdx.x == 0) partsv[blockIdx.x*8 + j] = sm[0];
    __syncthreads();
  }
}

__global__ void k_att4(const float* __restrict__ partse, const float* __restrict__ partsv,
                       const float* __restrict__ wr, const float* __restrict__ br,
                       float* __restrict__ addvec){
  int n = blockIdx.x;
  __shared__ float sm[256];
  __shared__ float ctx[8]; __shared__ float sse;
  sm[threadIdx.x] = partse[n*256 + threadIdx.x]; __syncthreads();
  for (int s = 128; s > 0; s >>= 1){
    if (threadIdx.x < s) sm[threadIdx.x] += sm[threadIdx.x + s];
    __syncthreads();
  }
  if (threadIdx.x == 0) sse = sm[0];
  __syncthreads();
  for (int j = 0; j < 8; ++j){
    sm[threadIdx.x] = partsv[(n*256 + threadIdx.x)*8 + j]; __syncthreads();
    for (int s = 128; s > 0; s >>= 1){
      if (threadIdx.x < s) sm[threadIdx.x] += sm[threadIdx.x + s];
      __syncthreads();
    }
    if (threadIdx.x == 0) ctx[j] = sm[0]/sse;
    __syncthreads();
  }
  if (threadIdx.x < 64){
    float a = br[threadIdx.x];
#pragma unroll
    for (int j = 0; j < 8; ++j) a += wr[threadIdx.x*8 + j]*ctx[j];
    addvec[n*64 + threadIdx.x] = a;
  }
}

// ================= transformer =================
__global__ void k_ln(const float* __restrict__ outp, const float* __restrict__ addvec,
                     const float* __restrict__ g, const float* __restrict__ bta,
                     _Float16* __restrict__ tn){
  int p = blockIdx.x*256 + threadIdx.x;
  int n = p >> 16;
  const float* t = outp + (size_t)p*64;
  const float* av = addvec + n*64;
  float buf[64]; float mu = 0.f;
#pragma unroll
  for (int c = 0; c < 64; ++c){ buf[c] = t[c] + av[c]; mu += buf[c]; }
  mu *= (1.f/64.f);
  float var = 0.f;
#pragma unroll
  for (int c = 0; c < 64; ++c){ float d = buf[c] - mu; var += d*d; }
  var *= (1.f/64.f);
  float inv = rsqrtf(var + 1e-5f);
  _Float16* q = tn + (size_t)p*64;
#pragma unroll
  for (int c = 0; c < 64; ++c) q[c] = (_Float16)((buf[c] - mu)*inv*g[c] + bta[c]);
}

// MLP with async global->LDS staging of the token tile (ASYNCcnt path).
__global__ void k_mlp(const _Float16* __restrict__ tn, const _Float16* __restrict__ w1,
                      const float* __restrict__ b1, const _Float16* __restrict__ w2,
                      const float* __restrict__ b2, const float* __restrict__ outp,
                      const float* __restrict__ addvec, float* __restrict__ dout){
  __shared__ __align__(16) _Float16 tbuf[8][16][64];   // staged tokens (16KB)
  __shared__ __align__(16) _Float16 hbuf[8][16][64];   // hidden (16KB)
  int warp = threadIdx.x >> 5;
  int lane = threadIdx.x & 31, hi = lane >> 4;
  int wid = (blockIdx.x*256 + threadIdx.x) >> 5;
  int pbase = wid << 4;

  // --- async copy: this wave's 16 tokens * 128B = 2KB, 16B/lane * 4 issues ---
  {
    const char* gbase = (const char*)(tn + (size_t)pbase*64);
    unsigned lbase = (unsigned)(uintptr_t)(&tbuf[warp][0][0]); // LDS offset = low 32 bits
#pragma unroll
    for (int it = 0; it < 4; ++it){
      unsigned boff = (unsigned)(it*512 + lane*16);
      unsigned long long gaddr = (unsigned long long)(uintptr_t)(gbase + boff);
      unsigned laddr = lbase + boff;
      asm volatile("global_load_async_to_lds_b128 %0, %1, off"
                   :: "v"(laddr), "v"(gaddr) : "memory");
    }
    asm volatile("s_wait_asynccnt 0x0" ::: "memory");
  }

  const _Float16* tp = &tbuf[warp][lane & 15][0];
  v8f acc[4] = {{},{},{},{}};
#pragma unroll
  for (int chunk = 0; chunk < 2; ++chunk){
    v16h a;
    ld8lo(a, tp + chunk*32 + hi*8);
    ld8hi(a, tp + chunk*32 + 16 + hi*8);
#pragma unroll
    for (int nt = 0; nt < 4; ++nt){
      v16h b = *(const v16h*)(w1 + ((nt*2 + chunk)*32 + lane)*16);
      acc[nt] = WMMA_F16(a, b, acc[nt]);
    }
  }
#pragma unroll
  for (int nt = 0; nt < 4; ++nt){
    int cc = nt*16 + (lane & 15);
    float bb = b1[cc];
#pragma unroll
    for (int r = 0; r < 8; ++r){
      float v = acc[nt][r] + bb;
      v = 0.5f*v*(1.f + erff(v*0.70710678118654752f));   // exact GELU
      hbuf[warp][hi*8 + r][cc] = (_Float16)v;
    }
  }
  __syncthreads();
  v8f acc2[4] = {{},{},{},{}};
  const _Float16* hp = &hbuf[warp][lane & 15][0];
#pragma unroll
  for (int chunk = 0; chunk < 2; ++chunk){
    v16h a;
    ld8lo(a, hp + chunk*32 + hi*8);
    ld8hi(a, hp + chunk*32 + 16 + hi*8);
#pragma unroll
    for (int nt = 0; nt < 4; ++nt){
      v16h b = *(const v16h*)(w2 + ((nt*2 + chunk)*32 + lane)*16);
      acc2[nt] = WMMA_F16(a, b, acc2[nt]);
    }
  }
  int n = pbase >> 16; int hw0 = pbase & 65535;
#pragma unroll
  for (int nt = 0; nt < 4; ++nt){
    int cc = nt*16 + (lane & 15);
    float bb = b2[cc]; float av = addvec[n*64 + cc];
#pragma unroll
    for (int r = 0; r < 8; ++r){
      int m = hi*8 + r;
      float tval = outp[(size_t)(pbase + m)*64 + cc] + av;
      dout[((size_t)(n*64 + cc) << 16) + hw0 + m] = acc2[nt][r] + bb + tval;
    }
  }
}

// ================= launch =================
extern "C" void kernel_launch(void* const* d_in, const int* in_sizes, int n_in,
                              void* d_out, int out_size, void* d_ws, size_t ws_size,
                              hipStream_t stream){
  (void)in_sizes; (void)n_in; (void)out_size; (void)ws_size;
  const float* x     = (const float*)d_in[0];
  const float* c7a_w = (const float*)d_in[1];  const float* c7a_b = (const float*)d_in[2];
  const float* c1_w  = (const float*)d_in[3];  const float* c1_b  = (const float*)d_in[4];
  const float* bn1_g = (const float*)d_in[5];  const float* bn1_b = (const float*)d_in[6];
  const float* rb1_w = (const float*)d_in[7];  const float* rb1_b = (const float*)d_in[8];
  const float* rbbn1_g = (const float*)d_in[9];  const float* rbbn1_b = (const float*)d_in[10];
  const float* rb2_w = (const float*)d_in[11]; const float* rb2_b = (const float*)d_in[12];
  const float* rbbn2_g = (const float*)d_in[13]; const float* rbbn2_b = (const float*)d_in[14];
  const float* c3_w  = (const float*)d_in[15]; const float* c3_b  = (const float*)d_in[16];
  const float* bn2_g = (const float*)d_in[17]; const float* bn2_b = (const float*)d_in[18];
  const float* c7b_w = (const float*)d_in[19]; const float* c7b_b = (const float*)d_in[20];
  const float* ak_w  = (const float*)d_in[21]; const float* ak_b  = (const float*)d_in[22];
  const float* av_w  = (const float*)d_in[25]; const float* av_b  = (const float*)d_in[26];
  const float* ar_w  = (const float*)d_in[27]; const float* ar_b  = (const float*)d_in[28];
  const float* ln_g  = (const float*)d_in[29]; const float* ln_b  = (const float*)d_in[30];
  const float* fc1_w = (const float*)d_in[31]; const float* fc1_b = (const float*)d_in[32];
  const float* fc2_w = (const float*)d_in[33]; const float* fc2_b = (const float*)d_in[34];

  char* ws = (char*)d_ws;
  _Float16* xh   = (_Float16*)(ws + OFF_XHALF);
  _Float16* y1   = (_Float16*)(ws + OFF_Y1);
  _Float16* y2   = (_Float16*)(ws + OFF_Y2);
  _Float16* y256 = (_Float16*)(ws + OFF_Y256);
  _Float16* z1   = (_Float16*)(ws + OFF_Z1);
  _Float16* z2   = (_Float16*)(ws + OFF_Z2);
  float*    outpre = (float*)(ws + OFF_OUTPRE);
  _Float16* tnb  = (_Float16*)(ws + OFF_TN);
  _Float16* w7a  = (_Float16*)(ws + OFF_W7A);
  _Float16* w1p  = (_Float16*)(ws + OFF_W1);
  _Float16* w3a  = (_Float16*)(ws + OFF_W3A);
  _Float16* w3b  = (_Float16*)(ws + OFF_W3B);
  _Float16* w3c  = (_Float16*)(ws + OFF_W3C);
  _Float16* w7b  = (_Float16*)(ws + OFF_W7B);
  _Float16* wfc1 = (_Float16*)(ws + OFF_WFC1);
  _Float16* wfc2 = (_Float16*)(ws + OFF_WFC2);
  float* pmax  = (float*)(ws + OFF_PMAX);
  float* kmaxp = (float*)(ws + OFF_KMAX);
  float* pse   = (float*)(ws + OFF_PSE);
  float* psv   = (float*)(ws + OFF_PSV);
  float* addv  = (float*)(ws + OFF_ADDV);

  // prep
  k_xhalf<<<32768, 256, 0, stream>>>(x, xh);
  k_pack7a<<<196, 256, 0, stream>>>(c7a_w, w7a);
  k_pack1<<<2, 256, 0, stream>>>(c1_w, w1p);
  k_pack3<<<10, 256, 0, stream>>>(rb1_w, w3a);
  k_pack3<<<10, 256, 0, stream>>>(rb2_w, w3b);
  k_pack3<<<10, 256, 0, stream>>>(c3_w, w3c);
  k_pack7b<<<200, 256, 0, stream>>>(c7b_w, w7b);
  k_packfc<<<16, 256, 0, stream>>>(fc1_w, wfc1);
  k_packfc<<<16, 256, 0, stream>>>(fc2_w, wfc2);
  // resizer
  k_conv7a<<<1024, 256, 0, stream>>>(xh, w7a, c7a_b, y1);
  k_conv1bn<<<1024, 256, 0, stream>>>(y1, w1p, c1_b, bn1_g, bn1_b, y2);
  k_resize16<<<2048, 256, 0, stream>>>(y2, y256);
  k_conv3<<<4096, 256, 0, stream>>>(y256, w3a, rb1_b, rbbn1_g, rbbn1_b, nullptr, 1, z1);
  k_conv3<<<4096, 256, 0, stream>>>(z1, w3b, rb2_b, rbbn2_g, rbbn2_b, y256, 0, z2);
  k_conv3<<<4096, 256, 0, stream>>>(z2, w3c, c3_b, bn2_g, bn2_b, y256, 0, z1);
  k_conv7b<<<16384, 256, 0, stream>>>(z1, w7b, c7b_b, x, outpre);
  // attention -> per-batch bias
  k_att1<<<2048, 256, 0, stream>>>(outpre, ak_w, ak_b, pmax);
  k_att2<<<8, 256, 0, stream>>>(pmax, kmaxp);
  k_att3<<<2048, 256, 0, stream>>>(outpre, ak_w, ak_b, av_w, av_b, kmaxp, pse, psv);
  k_att4<<<8, 256, 0, stream>>>(pse, psv, ar_w, ar_b, addv);
  // transformer
  k_ln<<<2048, 256, 0, stream>>>(outpre, addv, ln_g, ln_b, tnb);
  k_mlp<<<4096, 256, 0, stream>>>(tnb, wfc1, fc1_b, wfc2, fc2_b, outpre, addv, (float*)d_out);
}
